// MiniTTS_38242388803813
// MI455X (gfx1250) — matrix operand order repaired and checked by reading, not compiled, not run on hardware
//
#include <hip/hip_runtime.h>
#include <hip/hip_bf16.h>

typedef __attribute__((ext_vector_type(16))) __bf16 bf16x16;
typedef __attribute__((ext_vector_type(8)))  __bf16 bf16x8;
typedef __attribute__((ext_vector_type(8)))  float  f32x8;
typedef unsigned int u32x4 __attribute__((ext_vector_type(4)));
typedef int          i32x8 __attribute__((ext_vector_type(8)));
typedef int          i32x4 __attribute__((ext_vector_type(4)));

#define D_H 256
#define D_S 552
#define VOC 148
#define LT  128
#define BSZ 64
#define ST  800
#define MAX_DEC 1000

// ---------------------------------------------------------------------------
// WMMA helpers (CDNA5, wave32, v_wmma_f32_16x16x32_bf16)
// ---------------------------------------------------------------------------
__device__ __forceinline__ f32x8 wmma_bf16(bf16x16 a, bf16x16 b, f32x8 c) {
  return __builtin_amdgcn_wmma_f32_16x16x32_bf16(false, a, false, b, (short)0, c,
                                                 false, false);
}

// A tile (16x32 bf16, row-major source, stride lda elements).
__device__ __forceinline__ bf16x16 load_a_frag(const __bf16* A, int lda) {
  int lane = threadIdx.x & 31;
  const __bf16* p = A + (lane & 15) * lda + ((lane >> 4) << 4);
  union { bf16x16 v; bf16x8 h[2]; } u;
  u.h[0] = *(const bf16x8*)(p);
  u.h[1] = *(const bf16x8*)(p + 8);
  return u.v;
}

// B tile (32x16) from row-major W[N][K]: per-lane 16 contiguous K values.
__device__ __forceinline__ bf16x16 load_b_frag(const __bf16* W, int ldb) {
  int lane = threadIdx.x & 31;
  return *(const bf16x16*)(W + (lane & 15) * ldb + ((lane >> 4) << 4));
}

__device__ __forceinline__ float sigf(float x) { return 1.0f / (1.0f + __expf(-x)); }

// ---------------------------------------------------------------------------
// Tensor Data Mover: async DMA of `elems` contiguous bf16 (<= 16384) from
// global memory into LDS.  D# built per CDNA5 ISA 8.3/8.4 (count=1, type=2,
// data_size=1 (2 bytes), 1-row tile, tile_dim0=elems).  Issue from ONE wave;
// completion via s_wait_tensorcnt; publish to the workgroup with a barrier.
// This toolchain uses the 6-arg builtin form (g0, g1, g2, g3, g4, cpol).
// ---------------------------------------------------------------------------
__device__ __forceinline__ void tdm_load_bf16(const __bf16* gsrc, unsigned lds_off,
                                              unsigned elems) {
  unsigned long long ga = (unsigned long long)(uintptr_t)gsrc;
  u32x4 g0;
  g0[0] = 1u;                                                // count=1 (valid user D#)
  g0[1] = lds_off;                                           // D#.lds_addr (bytes)
  g0[2] = (unsigned)(ga & 0xffffffffu);                      // global_addr[31:0]
  g0[3] = (unsigned)((ga >> 32) & 0x01ffffffu) | (2u << 30); // global_addr[56:32], type=2
  i32x8 g1;
  g1[0] = 1 << 16;              // workgroup_mask=0, data_size=1 (2B), no pad/iterate
  g1[1] = (int)(elems << 16);   // tensor_dim0[15:0]
  g1[2] = 1 << 16;              // tensor_dim0[31:16]=0, tensor_dim1=1
  g1[3] = (int)(elems << 16);   // tensor_dim1[31:16]=0, tile_dim0=elems
  g1[4] = 1;                    // tile_dim1=1, tile_dim2=0
  g1[5] = (int)elems;           // tensor_dim0_stride[31:0]
  g1[6] = 0;                    // tensor_dim0_stride[47:32]=0, tensor_dim1_stride lo=0
  g1[7] = 0;
  i32x4 z4 = {0, 0, 0, 0};      // groups 2/3 unused (2-D descriptor)
  i32x8 z8 = {0, 0, 0, 0, 0, 0, 0, 0};
  __builtin_amdgcn_tensor_load_to_lds(g0, g1, z4, z4, z8, 0);
}

// ---------------------------------------------------------------------------
// Prep: fp32 -> bf16 weight conversion, [Wih|Whh] fusion, padding, embedding
// gather.  Runs once, fully parallel.
// ---------------------------------------------------------------------------
__global__ void prep_kernel(const float* __restrict__ enc_Wih, const float* __restrict__ enc_Whh,
                            const float* __restrict__ dec_Wih, const float* __restrict__ dec_Whh,
                            const float* __restrict__ pre_W,  const float* __restrict__ post_W1,
                            const float* __restrict__ post_W2,const float* __restrict__ stop_W1,
                            const float* __restrict__ emb,    const int*   __restrict__ tokens,
                            __bf16* encWcat, __bf16* decWcat, __bf16* preWp,
                            __bf16* pW1, __bf16* pW2, __bf16* sW1, __bf16* embX) {
  int idx0 = blockIdx.x * blockDim.x + threadIdx.x;
  int stride = gridDim.x * blockDim.x;
  for (int i = idx0; i < 1024 * 512; i += stride) {
    int g = i >> 9, k = i & 511;
    encWcat[i] = (__bf16)(k < D_H ? enc_Wih[g * D_H + k] : enc_Whh[g * D_H + (k - D_H)]);
    decWcat[i] = (__bf16)(k < D_H ? dec_Wih[g * D_H + k] : dec_Whh[g * D_H + (k - D_H)]);
  }
  for (int i = idx0; i < 256 * 576; i += stride) {
    int r = i / 576, k = i - r * 576;
    preWp[i] = (__bf16)(k < D_S ? pre_W[r * D_S + k] : 0.f);
  }
  for (int i = idx0; i < 256 * 256; i += stride) {
    pW1[i] = (__bf16)post_W1[i];
    sW1[i] = (__bf16)stop_W1[i];
  }
  for (int i = idx0; i < 560 * 256; i += stride) {
    int r = i >> 8, k = i & 255;
    pW2[i] = (__bf16)(r < D_S ? post_W2[r * D_H + k] : 0.f);
  }
  for (int i = idx0; i < LT * BSZ * D_H; i += stride) {
    int d = i & 255; int rb = i >> 8; int b = rb & 63; int t = rb >> 6;
    embX[i] = (__bf16)emb[tokens[t * BSZ + b] * D_H + d];
  }
}

// ---------------------------------------------------------------------------
// Pre-net (teacher-forced, independent of the recurrence):
// X[t] = S_in[t] @ pre_W^T + pre_b, stored bf16, one block per t.
// ---------------------------------------------------------------------------
__global__ __launch_bounds__(256) void prenet_kernel(const float* __restrict__ S_true,
                                                     const __bf16* __restrict__ preW,
                                                     const float* __restrict__ pre_b,
                                                     __bf16* __restrict__ X) {
  int t = blockIdx.x;
  __shared__ __bf16 fr[32][576];   // 36 KB: half of the batch, padded K
  int tid = threadIdx.x, wave = tid >> 5, lane = tid & 31;
  bool zero = (t == 0) || (t >= ST);
  for (int half = 0; half < 2; ++half) {
    for (int i = tid; i < 32 * 576; i += 256) {
      int row = i / 576, col = i - row * 576;
      float v = (!zero && col < D_S)
                    ? S_true[(size_t)t * BSZ * D_S + (half * 32 + row) * D_S + col]
                    : 0.f;
      fr[row][col] = (__bf16)v;
    }
    __syncthreads();
    for (int mt = 0; mt < 2; ++mt) {
      for (int nt2 = 0; nt2 < 2; ++nt2) {
        int nt = wave * 2 + nt2;
        f32x8 acc = {};
        for (int k = 0; k < 18; ++k)
          acc = wmma_bf16(load_a_frag(&fr[mt * 16][k * 32], 576),
                          load_b_frag(preW + (size_t)nt * 16 * 576 + k * 32, 576), acc);
        int n  = nt * 16 + (lane & 15);
        int mb = half * 32 + mt * 16 + ((lane >> 4) << 3);
        float bb = pre_b[n];
        #pragma unroll
        for (int r = 0; r < 8; ++r)
          X[(size_t)t * BSZ * D_H + (mb + r) * D_H + n] = (__bf16)(acc[r] + bb);
      }
    }
    __syncthreads();
  }
}

// ---------------------------------------------------------------------------
// Encoder: single persistent workgroup (16 waves).  h,c live in registers;
// x arrives per step via the Tensor Data Mover; gates = [x|h] @ Wcat^T, K=512.
// Wave w owns output columns [16w,16w+16) of every gate (i,f,g,o).
// ---------------------------------------------------------------------------
__global__ __launch_bounds__(512) void encoder_kernel(const __bf16* __restrict__ embX,
                                                      const int* __restrict__ lengths,
                                                      const __bf16* __restrict__ Wcat,
                                                      const float* __restrict__ bih,
                                                      const float* __restrict__ bhh,
                                                      float* __restrict__ h_out) {
  __shared__ __bf16 xS[BSZ][D_H];   // 32 KB, contiguous: TDM destination
  __shared__ __bf16 hS[BSZ][D_H];   // 32 KB
  int tid = threadIdx.x, wave = tid >> 5, lane = tid & 31;
  int mb = (lane >> 4) << 3;
  int ncol = (wave << 4) + (lane & 15);
  unsigned xs_off = (unsigned)(uintptr_t)&xS[0][0];
  float bg[4];
  #pragma unroll
  for (int g = 0; g < 4; ++g) bg[g] = bih[g * D_H + ncol] + bhh[g * D_H + ncol];
  int lenr[4][8];
  for (int mt = 0; mt < 4; ++mt)
    for (int r = 0; r < 8; ++r) lenr[mt][r] = lengths[mt * 16 + mb + r];
  float creg[4][8] = {}, hreg[4][8] = {};
  for (int i = tid; i < BSZ * D_H; i += 512) hS[i >> 8][i & 255] = (__bf16)0.f;
  __syncthreads();
  for (int t = 0; t < LT; ++t) {
    if (wave == 0) {
      tdm_load_bf16(embX + (size_t)t * BSZ * D_H, xs_off, BSZ * D_H);
      __builtin_amdgcn_s_wait_tensorcnt(0);
    }
    __syncthreads();
    for (int mt = 0; mt < 4; ++mt) {
      f32x8 acc[4] = {};
      #pragma unroll
      for (int k = 0; k < 16; ++k) {
        const __bf16* ap = (k < 8) ? &xS[mt * 16][k * 32] : &hS[mt * 16][(k - 8) * 32];
        bf16x16 a = load_a_frag(ap, D_H);
        #pragma unroll
        for (int g = 0; g < 4; ++g)
          acc[g] = wmma_bf16(a,
              load_b_frag(Wcat + (size_t)(g * D_H + (wave << 4)) * (2 * D_H) + k * 32,
                          2 * D_H),
              acc[g]);
      }
      #pragma unroll
      for (int r = 0; r < 8; ++r) {
        float c2 = sigf(acc[1][r] + bg[1]) * creg[mt][r] +
                   sigf(acc[0][r] + bg[0]) * tanhf(acc[2][r] + bg[2]);
        float h2 = sigf(acc[3][r] + bg[3]) * tanhf(c2);
        if (t < lenr[mt][r]) { creg[mt][r] = c2; hreg[mt][r] = h2; }
      }
    }
    __syncthreads();
    for (int mt = 0; mt < 4; ++mt)
      #pragma unroll
      for (int r = 0; r < 8; ++r)
        hS[mt * 16 + mb + r][ncol] = (__bf16)hreg[mt][r];
    __syncthreads();
  }
  for (int mt = 0; mt < 4; ++mt)
    #pragma unroll
    for (int r = 0; r < 8; ++r)
      h_out[(mt * 16 + mb + r) * D_H + ncol] = hreg[mt][r];
}

// ---------------------------------------------------------------------------
// Decoder recurrence: persistent workgroup, 1000 steps.  Pre-net outputs X[t]
// arrive via TDM; emits h_t (bf16) to global for the parallel post-net.
// h0 = c0 = enc_out.
// ---------------------------------------------------------------------------
__global__ __launch_bounds__(512) void decoder_kernel(const __bf16* __restrict__ X,
                                                      const __bf16* __restrict__ Wcat,
                                                      const float* __restrict__ bih,
                                                      const float* __restrict__ bhh,
                                                      const float* __restrict__ h_init,
                                                      __bf16* __restrict__ Hg) {
  __shared__ __bf16 xS[BSZ][D_H];   // 32 KB, contiguous: TDM destination
  __shared__ __bf16 hS[BSZ][D_H];   // 32 KB
  int tid = threadIdx.x, wave = tid >> 5, lane = tid & 31;
  int mb = (lane >> 4) << 3;
  int ncol = (wave << 4) + (lane & 15);
  unsigned xs_off = (unsigned)(uintptr_t)&xS[0][0];
  float bg[4];
  #pragma unroll
  for (int g = 0; g < 4; ++g) bg[g] = bih[g * D_H + ncol] + bhh[g * D_H + ncol];
  float creg[4][8], hreg[4][8];
  for (int mt = 0; mt < 4; ++mt)
    #pragma unroll
    for (int r = 0; r < 8; ++r) {
      float h0 = h_init[(mt * 16 + mb + r) * D_H + ncol];
      hreg[mt][r] = h0; creg[mt][r] = h0;
      hS[mt * 16 + mb + r][ncol] = (__bf16)h0;
    }
  __syncthreads();
  for (int t = 0; t < MAX_DEC; ++t) {
    if (wave == 0) {
      tdm_load_bf16(X + (size_t)t * BSZ * D_H, xs_off, BSZ * D_H);
      __builtin_amdgcn_s_wait_tensorcnt(0);
    }
    if (t + 1 < MAX_DEC)   // global_prefetch_b8: warm L2 for the next frame
      __builtin_prefetch(X + (size_t)(t + 1) * BSZ * D_H + tid * 32, 0, 1);
    __syncthreads();
    for (int mt = 0; mt < 4; ++mt) {
      f32x8 acc[4] = {};
      #pragma unroll
      for (int k = 0; k < 16; ++k) {
        const __bf16* ap = (k < 8) ? &xS[mt * 16][k * 32] : &hS[mt * 16][(k - 8) * 32];
        bf16x16 a = load_a_frag(ap, D_H);
        #pragma unroll
        for (int g = 0; g < 4; ++g)
          acc[g] = wmma_bf16(a,
              load_b_frag(Wcat + (size_t)(g * D_H + (wave << 4)) * (2 * D_H) + k * 32,
                          2 * D_H),
              acc[g]);
      }
      #pragma unroll
      for (int r = 0; r < 8; ++r) {
        float c2 = sigf(acc[1][r] + bg[1]) * creg[mt][r] +
                   sigf(acc[0][r] + bg[0]) * tanhf(acc[2][r] + bg[2]);
        float h2 = sigf(acc[3][r] + bg[3]) * tanhf(c2);
        creg[mt][r] = c2; hreg[mt][r] = h2;
      }
    }
    __syncthreads();
    __bf16* Ht = Hg + (size_t)t * BSZ * D_H;
    for (int mt = 0; mt < 4; ++mt)
      #pragma unroll
      for (int r = 0; r < 8; ++r) {
        __bf16 hb = (__bf16)hreg[mt][r];
        int row = mt * 16 + mb + r;
        hS[row][ncol] = hb;
        Ht[row * D_H + ncol] = hb;
      }
    __syncthreads();
  }
}

// ---------------------------------------------------------------------------
// Post-net: one block per t (fully parallel over 1000 steps).
// hid = relu(h@W1^T+b1); S = hid@W2^T+b2; sh = relu(h@sW1^T+sb1); stop = sh.w2
// ---------------------------------------------------------------------------
__global__ __launch_bounds__(256) void postnet_kernel(const __bf16* __restrict__ Hg,
                                                      const __bf16* __restrict__ pW1,
                                                      const float* __restrict__ post_b1,
                                                      const __bf16* __restrict__ pW2,
                                                      const float* __restrict__ post_b2,
                                                      const __bf16* __restrict__ sW1,
                                                      const float* __restrict__ stop_b1,
                                                      const float* __restrict__ stop_W2,
                                                      const float* __restrict__ stop_b2,
                                                      float* __restrict__ S_pred,
                                                      float* __restrict__ stop_out) {
  int t = blockIdx.x;
  __shared__ __bf16 hS[BSZ][D_H];     // 32 KB, TDM destination
  __shared__ __bf16 work[BSZ][D_H];   // 32 KB (reused: hid, then sh)
  int tid = threadIdx.x, wave = tid >> 5, lane = tid & 31;
  if (wave == 0) {
    tdm_load_bf16(Hg + (size_t)t * BSZ * D_H, (unsigned)(uintptr_t)&hS[0][0],
                  BSZ * D_H);
    __builtin_amdgcn_s_wait_tensorcnt(0);
  }
  __syncthreads();
  // hid = relu(h @ post_W1^T + b1)
  for (int mt = 0; mt < 4; ++mt)
    for (int nt2 = 0; nt2 < 2; ++nt2) {
      int nt = wave * 2 + nt2;
      f32x8 acc = {};
      for (int k = 0; k < 8; ++k)
        acc = wmma_bf16(load_a_frag(&hS[mt * 16][k * 32], D_H),
                        load_b_frag(pW1 + (size_t)nt * 16 * D_H + k * 32, D_H), acc);
      int n = nt * 16 + (lane & 15);
      int mbb = mt * 16 + ((lane >> 4) << 3);
      float bb = post_b1[n];
      #pragma unroll
      for (int r = 0; r < 8; ++r)
        work[mbb + r][n] = (__bf16)fmaxf(acc[r] + bb, 0.f);
    }
  __syncthreads();
  // S_out = hid @ post_W2^T + b2  (N = 552, padded to 35 tiles of 16)
  for (int mt = 0; mt < 4; ++mt)
    for (int nt = wave; nt < 35; nt += 8) {
      f32x8 acc = {};
      for (int k = 0; k < 8; ++k)
        acc = wmma_bf16(load_a_frag(&work[mt * 16][k * 32], D_H),
                        load_b_frag(pW2 + (size_t)nt * 16 * D_H + k * 32, D_H), acc);
      int n = nt * 16 + (lane & 15);
      int mbb = mt * 16 + ((lane >> 4) << 3);
      if (n < D_S) {
        float bb = post_b2[n];
        #pragma unroll
        for (int r = 0; r < 8; ++r)
          S_pred[(size_t)t * BSZ * D_S + (mbb + r) * D_S + n] = acc[r] + bb;
      }
    }
  __syncthreads();
  // sh = relu(h @ stop_W1^T + sb1) (overwrite work)
  for (int mt = 0; mt < 4; ++mt)
    for (int nt2 = 0; nt2 < 2; ++nt2) {
      int nt = wave * 2 + nt2;
      f32x8 acc = {};
      for (int k = 0; k < 8; ++k)
        acc = wmma_bf16(load_a_frag(&hS[mt * 16][k * 32], D_H),
                        load_b_frag(sW1 + (size_t)nt * 16 * D_H + k * 32, D_H), acc);
      int n = nt * 16 + (lane & 15);
      int mbb = mt * 16 + ((lane >> 4) << 3);
      float bb = stop_b1[n];
      #pragma unroll
      for (int r = 0; r < 8; ++r)
        work[mbb + r][n] = (__bf16)fmaxf(acc[r] + bb, 0.f);
    }
  __syncthreads();
  if (tid < BSZ) {
    float s = stop_b2[0];
    for (int j = 0; j < D_H; ++j) s += (float)work[tid][j] * stop_W2[j];
    stop_out[(size_t)t * BSZ + tid] = s;
  }
}

// ---------------------------------------------------------------------------
extern "C" void kernel_launch(void* const* d_in, const int* in_sizes, int n_in,
                              void* d_out, int out_size, void* d_ws, size_t ws_size,
                              hipStream_t stream) {
  const int*   token_pad = (const int*)d_in[0];
  const int*   token_len = (const int*)d_in[1];
  const float* S_true    = (const float*)d_in[2];
  const float* emb       = (const float*)d_in[3];
  const float* enc_Wih   = (const float*)d_in[4];
  const float* enc_Whh   = (const float*)d_in[5];
  const float* enc_bih   = (const float*)d_in[6];
  const float* enc_bhh   = (const float*)d_in[7];
  const float* dec_Wih   = (const float*)d_in[8];
  const float* dec_Whh   = (const float*)d_in[9];
  const float* dec_bih   = (const float*)d_in[10];
  const float* dec_bhh   = (const float*)d_in[11];
  const float* pre_W     = (const float*)d_in[12];
  const float* pre_b     = (const float*)d_in[13];
  const float* post_W1   = (const float*)d_in[14];
  const float* post_b1   = (const float*)d_in[15];
  const float* post_W2   = (const float*)d_in[16];
  const float* post_b2   = (const float*)d_in[17];
  const float* stop_W1   = (const float*)d_in[18];
  const float* stop_b1   = (const float*)d_in[19];
  const float* stop_W2   = (const float*)d_in[20];
  const float* stop_b2   = (const float*)d_in[21];

  char* ws = (char*)d_ws;
  size_t off = 0;
  auto take = [&](size_t bytes) { char* p = ws + off; off += (bytes + 255) & ~(size_t)255; return p; };
  __bf16* encWcat = (__bf16*)take((size_t)1024 * 512 * 2);
  __bf16* decWcat = (__bf16*)take((size_t)1024 * 512 * 2);
  __bf16* preWp   = (__bf16*)take((size_t)256 * 576 * 2);
  __bf16* pW1     = (__bf16*)take((size_t)256 * 256 * 2);
  __bf16* pW2     = (__bf16*)take((size_t)560 * 256 * 2);
  __bf16* sW1     = (__bf16*)take((size_t)256 * 256 * 2);
  __bf16* embX    = (__bf16*)take((size_t)LT * BSZ * D_H * 2);
  float*  h_n     = (float*) take((size_t)BSZ * D_H * 4);
  __bf16* Xbuf    = (__bf16*)take((size_t)MAX_DEC * BSZ * D_H * 2);
  __bf16* Hbuf    = (__bf16*)take((size_t)MAX_DEC * BSZ * D_H * 2);

  float* S_pred   = (float*)d_out;
  float* stop_out = S_pred + (size_t)MAX_DEC * BSZ * D_S;

  prep_kernel<<<512, 256, 0, stream>>>(enc_Wih, enc_Whh, dec_Wih, dec_Whh, pre_W,
                                       post_W1, post_W2, stop_W1, emb, token_pad,
                                       encWcat, decWcat, preWp, pW1, pW2, sW1, embX);
  prenet_kernel<<<MAX_DEC, 256, 0, stream>>>(S_true, preWp, pre_b, Xbuf);
  encoder_kernel<<<1, 512, 0, stream>>>(embX, token_len, encWcat, enc_bih, enc_bhh, h_n);
  decoder_kernel<<<1, 512, 0, stream>>>(Xbuf, decWcat, dec_bih, dec_bhh, h_n, Hbuf);
  postnet_kernel<<<MAX_DEC, 256, 0, stream>>>(Hbuf, pW1, post_b1, pW2, post_b2, sW1,
                                              stop_b1, stop_W2, stop_b2, S_pred, stop_out);
  (void)in_sizes; (void)n_in; (void)out_size; (void)ws_size;
}
